// WindowCrossAttention_32719060861237
// MI455X (gfx1250) — matrix operand order repaired and checked by reading, not compiled, not run on hardware
//
#include <hip/hip_runtime.h>
#include <hip/hip_bf16.h>

typedef _Float16 h16;
typedef __attribute__((ext_vector_type(16))) _Float16 v16h;
typedef __attribute__((ext_vector_type(8)))  _Float16 v8h;
typedef __attribute__((ext_vector_type(8)))  float    v8f;
typedef __attribute__((ext_vector_type(4)))  float    v4f;

#define C_DIM   256
#define N_TOK   49
#define D_HEAD  32
#define NP      64
#define LN_EPSF 1e-5f
#define QSCALE  0.17677669529663687f  /* 1/sqrt(32) */

static __device__ __forceinline__ v8f wmma16(v16h a, v16h b, v8f c) {
  return __builtin_amdgcn_wmma_f32_16x16x32_f16(false, a, false, b, (short)0, c, false, false);
}

static __device__ __forceinline__ v8f splat8(float x) {
  v8f r;
  #pragma unroll
  for (int i = 0; i < 8; ++i) r[i] = x;
  return r;
}

// A-fragment (16x32 f16) from LDS, ISA 7.12.2 layout:
// lane l: row m = l%16 ; elems 0-7 -> K = (l/16)*8 + 0..7 ; elems 8-15 -> K = 16 + (l/16)*8 + 0..7
static __device__ __forceinline__ v16h load_a_lds(const h16* buf, int row, int ld, int kbase, int lane) {
  const h16* p = buf + row * ld + kbase + (lane >> 4) * 8;
  v8h lo = *(const v8h*)(p);
  v8h hi = *(const v8h*)(p + 16);
  v16h a;
  #pragma unroll
  for (int i = 0; i < 8; ++i) { a[i] = lo[i]; a[8 + i] = hi[i]; }
  return a;
}

// A-fragment from global f32 (row stride 256), converting to f16; rows >= nrows are zero.
static __device__ __forceinline__ v16h load_a_f32g(const float* buf, int row, int nrows, int kbase, int lane) {
  v16h a;
  if (row < nrows) {
    const float* p = buf + row * C_DIM + kbase + (lane >> 4) * 8;
    v4f f0 = *(const v4f*)(p);
    v4f f1 = *(const v4f*)(p + 4);
    v4f f2 = *(const v4f*)(p + 16);
    v4f f3 = *(const v4f*)(p + 20);
    #pragma unroll
    for (int i = 0; i < 4; ++i) {
      a[i]      = (h16)f0[i];
      a[4 + i]  = (h16)f1[i];
      a[8 + i]  = (h16)f2[i];
      a[12 + i] = (h16)f3[i];
    }
  } else {
    #pragma unroll
    for (int i = 0; i < 16; ++i) a[i] = (h16)0.f;
  }
  return a;
}

// ---------------- prep kernels ----------------

// wT[n][k] = (f16) w[n][k]  for the 4 weight matrices (B-fragment friendly: row = N dim)
__global__ void prep_weights_k(const float* __restrict__ wq, const float* __restrict__ wk,
                               const float* __restrict__ wv, const float* __restrict__ wp,
                               h16* __restrict__ wqT, h16* __restrict__ wkT,
                               h16* __restrict__ wvT, h16* __restrict__ wpT) {
  int k = blockIdx.x, n = threadIdx.x;
  int src = n * C_DIM + k, dst = n * C_DIM + k;
  wqT[dst] = (h16)wq[src];
  wkT[dst] = (h16)wk[src];
  wvT[dst] = (h16)wv[src];
  wpT[dst] = (h16)wp[src];
}

// biasp[h][n][m] (64x64 padded, 0 outside 49x49)
__global__ void prep_bias_k(const float* __restrict__ bt, const int* __restrict__ rel,
                            float* __restrict__ biasp) {
  int n = blockIdx.x, m = threadIdx.x;
  bool valid = (n < N_TOK) && (m < N_TOK);
  int idx = valid ? rel[n * N_TOK + m] : 0;
  #pragma unroll
  for (int h = 0; h < 8; ++h)
    biasp[((size_t)h * NP + n) * NP + m] = valid ? bt[idx * 8 + h] : 0.f;
}

// maskp[w][n][m] (64x64 padded per window)
__global__ void prep_mask_k(const float* __restrict__ sm, float* __restrict__ maskp) {
  int wn = blockIdx.x;           // w*64 + n
  int w = wn >> 6, n = wn & 63, m = threadIdx.x;
  maskp[(size_t)wn * NP + m] = ((n < N_TOK) && (m < N_TOK)) ? sm[((size_t)w * N_TOK + n) * N_TOK + m] : 0.f;
}

// ---------------- fused attention kernel ----------------
// grid = 4096 blocks (one window), 256 threads = 8 waves = 8 heads.

__global__ __launch_bounds__(256)
void swin_xattn_kernel(const float* __restrict__ x, const float* __restrict__ y,
                       const float* __restrict__ wq_b, const float* __restrict__ wk_b,
                       const float* __restrict__ wv_b, const float* __restrict__ proj_b,
                       const float* __restrict__ ln_g, const float* __restrict__ ln_b,
                       const h16* __restrict__ wqT, const h16* __restrict__ wkT,
                       const h16* __restrict__ wvT, const h16* __restrict__ wpT,
                       const float* __restrict__ biasp, const float* __restrict__ maskp,
                       float* __restrict__ out) {
  __shared__ h16   scr[8][1536];        // 3KB per-wave bounce: [0,512) K-transpose, [512,1536) V
  __shared__ h16   osd[NP * C_DIM];     // 32KB: attention output, f16, all heads
  __shared__ float lnbuf[NP][4];        // cross-wave LN partials {s1,s2} x 2 col-halves

  const int tid  = threadIdx.x;
  const int wave = tid >> 5;
  const int lane = tid & 31;
  const int l16  = lane & 15;
  const int lh   = lane >> 4;
  const int h    = wave;                // head == wave
  const int b    = blockIdx.x;
  const int wb   = b & 63;

  const float* xb = x + (size_t)b * N_TOK * C_DIM;
  const float* yb = y + (size_t)b * N_TOK * C_DIM;
  h16* myscr  = scr[wave];
  h16* vstage = myscr + 512;

  v16h qa[4], kbf[4], vbf[2][2];

  // ===== Q (head slice, kept as A-fragments; scale folded in) =====
  #pragma unroll
  for (int ti = 0; ti < 4; ++ti) {
    v8f acc[2];
    #pragma unroll
    for (int tn = 0; tn < 2; ++tn) acc[tn] = splat8(wq_b[h * D_HEAD + tn * 16 + l16]);
    #pragma unroll
    for (int kc = 0; kc < 8; ++kc) {
      v16h a = load_a_f32g(xb, ti * 16 + l16, N_TOK, kc * 32, lane);
      #pragma unroll
      for (int tn = 0; tn < 2; ++tn) {
        v16h bf = load_a_lds(wqT, h * D_HEAD + tn * 16 + l16, C_DIM, kc * 32, lane);
        acc[tn] = wmma16(a, bf, acc[tn]);
      }
    }
    #pragma unroll
    for (int tn = 0; tn < 2; ++tn)
      #pragma unroll
      for (int v = 0; v < 8; ++v)
        myscr[(v + 8 * lh) * 32 + tn * 16 + l16] = (h16)(acc[tn][v] * QSCALE);   // [16 tok][32 ch]
    __builtin_amdgcn_wave_barrier();
    qa[ti] = load_a_lds(myscr, l16, 32, 0, lane);
    __builtin_amdgcn_wave_barrier();
  }

  // ===== K and V fused: one pass over y, two WMMA chains per A-fragment =====
  #pragma unroll
  for (int kt = 0; kt < 2; ++kt) {
    #pragma unroll
    for (int sub = 0; sub < 2; ++sub) {
      int ti = kt * 2 + sub;
      v8f kacc[2], vacc[2];
      #pragma unroll
      for (int tn = 0; tn < 2; ++tn) {
        kacc[tn] = splat8(wk_b[h * D_HEAD + tn * 16 + l16]);
        vacc[tn] = splat8(wv_b[h * D_HEAD + tn * 16 + l16]);
      }
      #pragma unroll
      for (int kc = 0; kc < 8; ++kc) {
        v16h a = load_a_f32g(yb, ti * 16 + l16, N_TOK, kc * 32, lane);   // shared A
        #pragma unroll
        for (int tn = 0; tn < 2; ++tn) {
          v16h bk = load_a_lds(wkT, h * D_HEAD + tn * 16 + l16, C_DIM, kc * 32, lane);
          kacc[tn] = wmma16(a, bk, kacc[tn]);
          v16h bv = load_a_lds(wvT, h * D_HEAD + tn * 16 + l16, C_DIM, kc * 32, lane);
          vacc[tn] = wmma16(a, bv, vacc[tn]);
        }
      }
      // stage K transposed -> B-fragment of K^T
      #pragma unroll
      for (int tn = 0; tn < 2; ++tn)
        #pragma unroll
        for (int v = 0; v < 8; ++v)
          myscr[(v + 8 * lh) * 32 + tn * 16 + l16] = (h16)kacc[tn][v];         // [16 tok][32 ch]
      __builtin_amdgcn_wave_barrier();
      kbf[ti] = load_a_lds(myscr, l16, 32, 0, lane);    // B: lane = token col, elems = channels (table order)
      __builtin_amdgcn_wave_barrier();
      // stage V (persistent across the two sub-tiles of this 32-token chunk)
      #pragma unroll
      for (int tn = 0; tn < 2; ++tn)
        #pragma unroll
        for (int v = 0; v < 8; ++v)
          vstage[(tn * 16 + l16) * 32 + sub * 16 + v + 8 * lh] = (h16)vacc[tn][v]; // [32 ch][32 tok]
    }
    __builtin_amdgcn_wave_barrier();
    #pragma unroll
    for (int nt = 0; nt < 2; ++nt)
      vbf[kt][nt] = load_a_lds(vstage, nt * 16 + l16, 32, 0, lane);  // B: lane = channel col, elems = tokens
    __builtin_amdgcn_wave_barrier();
  }

  // ===== scores: S = Q K^T * scale + bias + mask  (C folded in) =====
  v8f S[4][4];
  const float* bh = biasp + (size_t)h * NP * NP;
  const float* mw = maskp + (size_t)wb * NP * NP;
  #pragma unroll
  for (int ti = 0; ti < 4; ++ti)
    #pragma unroll
    for (int tj = 0; tj < 4; ++tj) {
      v8f c;
      #pragma unroll
      for (int v = 0; v < 8; ++v) {
        int row = ti * 16 + v + 8 * lh, col = tj * 16 + l16;
        c[v] = bh[row * NP + col] + mw[row * NP + col];
      }
      S[ti][tj] = wmma16(qa[ti], kbf[tj], c);
    }

  // ===== softmax over 49 key columns (wave32 xor-shuffles; halves hold different rows) =====
  #pragma unroll
  for (int ti = 0; ti < 4; ++ti) {
    float mx[8], sm[8];
    #pragma unroll
    for (int v = 0; v < 8; ++v) {
      float m = -1e30f;
      #pragma unroll
      for (int tj = 0; tj < 4; ++tj) {
        float s = S[ti][tj][v];
        if (tj * 16 + l16 >= N_TOK) s = -1e30f;
        S[ti][tj][v] = s;
        m = fmaxf(m, s);
      }
      mx[v] = m;
    }
    #pragma unroll
    for (int d = 1; d < 16; d <<= 1)
      #pragma unroll
      for (int v = 0; v < 8; ++v) mx[v] = fmaxf(mx[v], __shfl_xor(mx[v], d, 32));
    #pragma unroll
    for (int v = 0; v < 8; ++v) {
      float s = 0.f;
      #pragma unroll
      for (int tj = 0; tj < 4; ++tj) {
        float p = __expf(S[ti][tj][v] - mx[v]);
        S[ti][tj][v] = p;
        s += p;
      }
      sm[v] = s;
    }
    #pragma unroll
    for (int d = 1; d < 16; d <<= 1)
      #pragma unroll
      for (int v = 0; v < 8; ++v) sm[v] += __shfl_xor(sm[v], d, 32);
    #pragma unroll
    for (int v = 0; v < 8; ++v) {
      float inv = 1.f / sm[v];
      #pragma unroll
      for (int tj = 0; tj < 4; ++tj) S[ti][tj][v] *= inv;
    }
  }

  // ===== O = P V  (P bounced per row-tile through LDS to A-layout) =====
  v8f O[4][2];
  #pragma unroll
  for (int ti = 0; ti < 4; ++ti) {
    #pragma unroll
    for (int tj = 0; tj < 4; ++tj)
      #pragma unroll
      for (int v = 0; v < 8; ++v)
        myscr[(v + 8 * lh) * 64 + tj * 16 + l16] = (h16)S[ti][tj][v];            // [16 tok][64 key]
    __builtin_amdgcn_wave_barrier();
    v16h pa0 = load_a_lds(myscr, l16, 64, 0, lane);
    v16h pa1 = load_a_lds(myscr, l16, 64, 32, lane);
    __builtin_amdgcn_wave_barrier();
    #pragma unroll
    for (int nt = 0; nt < 2; ++nt) {
      v8f acc = splat8(0.f);
      acc = wmma16(pa0, vbf[0][nt], acc);
      acc = wmma16(pa1, vbf[1][nt], acc);
      O[ti][nt] = acc;
    }
  }

  // stage O (all heads) into LDS for the projection
  #pragma unroll
  for (int ti = 0; ti < 4; ++ti)
    #pragma unroll
    for (int nt = 0; nt < 2; ++nt)
      #pragma unroll
      for (int v = 0; v < 8; ++v)
        osd[(ti * 16 + v + 8 * lh) * C_DIM + h * D_HEAD + nt * 16 + l16] = (h16)O[ti][nt][v];
  __syncthreads();

  // ===== projection + residual (C = proj_b + x) =====
  const int pti = wave & 3;            // row-tile
  const int tnb = (wave >> 2) * 8;     // column-tile base (two col-halves across wave groups)
  v8f P[8];
  #pragma unroll
  for (int t8 = 0; t8 < 8; ++t8) {
    int co = (tnb + t8) * 16 + l16;
    #pragma unroll
    for (int v = 0; v < 8; ++v) {
      int tok = pti * 16 + v + 8 * lh;
      P[t8][v] = proj_b[co] + (tok < N_TOK ? xb[tok * C_DIM + co] : 0.f);
    }
  }
  #pragma unroll
  for (int kc = 0; kc < 8; ++kc) {
    v16h a = load_a_lds(osd, pti * 16 + l16, C_DIM, kc * 32, lane);
    #pragma unroll
    for (int t8 = 0; t8 < 8; ++t8) {
      v16h bf = load_a_lds(wpT, (tnb + t8) * 16 + l16, C_DIM, kc * 32, lane);
      P[t8] = wmma16(a, bf, P[t8]);
    }
  }

  // ===== LayerNorm: in-register partials + 1KB cross-wave reduce =====
  #pragma unroll
  for (int v = 0; v < 8; ++v) {
    float s1 = 0.f, s2 = 0.f;
    #pragma unroll
    for (int t8 = 0; t8 < 8; ++t8) { float t = P[t8][v]; s1 += t; s2 += t * t; }
    #pragma unroll
    for (int d = 1; d < 16; d <<= 1) { s1 += __shfl_xor(s1, d, 32); s2 += __shfl_xor(s2, d, 32); }
    if (l16 == 0) {
      int row = pti * 16 + v + 8 * lh;
      lnbuf[row][(wave >> 2) * 2 + 0] = s1;
      lnbuf[row][(wave >> 2) * 2 + 1] = s2;
    }
  }
  __syncthreads();
  float mu[8], rs[8];
  #pragma unroll
  for (int v = 0; v < 8; ++v) {
    int row = pti * 16 + v + 8 * lh;
    float s1 = lnbuf[row][0] + lnbuf[row][2];
    float s2 = lnbuf[row][1] + lnbuf[row][3];
    float m  = s1 * (1.f / C_DIM);
    float var = s2 * (1.f / C_DIM) - m * m;
    mu[v] = m;
    rs[v] = rsqrtf(var + LN_EPSF);
  }
  #pragma unroll
  for (int t8 = 0; t8 < 8; ++t8) {
    int co = (tnb + t8) * 16 + l16;
    float g = ln_g[co], be = ln_b[co];
    #pragma unroll
    for (int v = 0; v < 8; ++v) {
      int tok = pti * 16 + v + 8 * lh;
      if (tok < N_TOK)
        out[((size_t)b * N_TOK + tok) * C_DIM + co] = (P[t8][v] - mu[v]) * rs[v] * g + be;
    }
  }
}

// ---------------- launch ----------------

extern "C" void kernel_launch(void* const* d_in, const int* in_sizes, int n_in,
                              void* d_out, int out_size, void* d_ws, size_t ws_size,
                              hipStream_t stream) {
  const float* x          = (const float*)d_in[0];
  const float* y          = (const float*)d_in[1];
  const float* shift_mask = (const float*)d_in[2];
  const float* wq_w       = (const float*)d_in[3];
  const float* wq_b       = (const float*)d_in[4];
  const float* wk_w       = (const float*)d_in[5];
  const float* wk_b       = (const float*)d_in[6];
  const float* wv_w       = (const float*)d_in[7];
  const float* wv_b       = (const float*)d_in[8];
  const float* bias_table = (const float*)d_in[9];
  const float* proj_w     = (const float*)d_in[10];
  const float* proj_b     = (const float*)d_in[11];
  const float* ln_g       = (const float*)d_in[12];
  const float* ln_b       = (const float*)d_in[13];
  const int*   rel_idx    = (const int*)d_in[14];

  char* ws = (char*)d_ws;
  h16*   wqT   = (h16*)(ws + 0);
  h16*   wkT   = (h16*)(ws + 131072);
  h16*   wvT   = (h16*)(ws + 262144);
  h16*   wpT   = (h16*)(ws + 393216);
  float* biasp = (float*)(ws + 524288);   // 8*64*64 f32  = 128KB
  float* maskp = (float*)(ws + 655360);   // 64*64*64 f32 = 1MB

  prep_weights_k<<<256, 256, 0, stream>>>(wq_w, wk_w, wv_w, proj_w, wqT, wkT, wvT, wpT);
  prep_bias_k<<<64, 64, 0, stream>>>(bias_table, rel_idx, biasp);
  prep_mask_k<<<4096, 64, 0, stream>>>(shift_mask, maskp);

  swin_xattn_kernel<<<4096, 256, 0, stream>>>(x, y, wq_b, wk_b, wv_b, proj_b, ln_g, ln_b,
                                              wqT, wkT, wvT, wpT, biasp, maskp,
                                              (float*)d_out);
}
